// lstm_decoder_42966852829695
// MI455X (gfx1250) — matrix operand order repaired and checked by reading, not compiled, not run on hardware
//
#include <hip/hip_runtime.h>
#include <hip/hip_bf16.h>
#include <stdint.h>

// LSTM autoregressive decoder for MI455X (gfx1250, wave32, WMMA).
//
// Restructuring:
//   gates_t = Gx[t] + h_t @ Weff.T (+ bias2, t>=1)     Weff.T = W_hh.T + W_lin.T @ Wp.T
//   Gx[t]   = x_t @ W_ih[:,F:].T + b_ih + b_hh          (parallel WMMA GEMM, frag-order f16)
//   out_t   = h_all[t] @ W_lin.T + b_lin                (parallel WMMA GEMM at the end)
// Serial kernel: 16 batch-tiles x 2 N-split WGs, weights LDS-resident (256KB/WG,
// staged with GLOBAL_LOAD_ASYNC_TO_LDS_B128), per-pair h exchange via L2 +
// release/acquire flags. Streaming buffers use non-temporal TH hints.

#define H_      256
#define F_      64
#define L_      512
#define B_      256

typedef _Float16 f16x16 __attribute__((ext_vector_type(16)));
typedef float    f32x8  __attribute__((ext_vector_type(8)));
typedef unsigned int u32x4v __attribute__((ext_vector_type(4)));

union Frag32 { f16x16 v; _Float16 h[16]; uint4 q[2]; u32x4v p[2]; };
union Half8  { _Float16 h[8]; uint4 q; u32x4v p; };

// workspace layout (bytes)
#define GX_OFF    ((size_t)0)                      // 8192*64 frags * 512B = 256MB (f16, C-frag order)
#define HALL_OFF  (GX_OFF   + (size_t)268435456)   // h_all [t][b][k] f16 = 64MB
#define BEFF_OFF  (HALL_OFF + (size_t)67108864)    // Weff.T frags, [s][w][g][kt] (contig 256KB/WG)
#define BHH_OFF   (BEFF_OFF + (size_t)524288)      // W_hh.T packed B-frags  [8][64] * 1KB
#define BX_OFF    (BHH_OFF  + (size_t)524288)      // Wx.T   packed B-frags  [8][64] * 1KB
#define BL_OFF    (BX_OFF   + (size_t)524288)      // Wl.T   packed B-frags  [8][4]  * 1KB
#define BSUM_OFF  (BL_OFF   + (size_t)32768)       // b_ih + b_hh       [1024] f32
#define BIAS2_OFF (BSUM_OFF + (size_t)4096)        // b_lin @ Wp.T      [1024] f32
#define FLAGS_OFF (BIAS2_OFF+ (size_t)4096)        // sync flags [16][2] u32

#define HPAD 264           // hstage row stride in halfs (528B: conflict-friendly, 16B aligned)
#define REC_LDS (262144 + 16*HPAD*2)

__device__ __forceinline__ float sigm_(float x) { return 1.0f / (1.0f + __expf(-x)); }
__device__ __forceinline__ float tanh_(float x) {
  x = fminf(fmaxf(x, -15.f), 15.f);
  float e = __expf(2.0f * x);
  return (e - 1.0f) / (e + 1.0f);
}

// ---- biases + flags ------------------------------------------------------
__global__ void k_prep(const float* __restrict__ b_ih, const float* __restrict__ b_hh,
                       const float* __restrict__ b_lin, const float* __restrict__ W_ih,
                       char* __restrict__ ws) {
  int n = blockIdx.x * 256 + threadIdx.x;
  if (n < 32) ((unsigned*)(ws + FLAGS_OFF))[n] = 0u;
  if (n < 1024) {
    ((float*)(ws + BSUM_OFF))[n] = b_ih[n] + b_hh[n];
    float d = 0.f;
    for (int f = 0; f < F_; ++f) d += b_lin[f] * W_ih[(size_t)n * 320 + f];
    ((float*)(ws + BIAS2_OFF))[n] = d;
  }
}

// ---- pack Weff.T and W_hh.T into B-fragment order ------------------------
// B-frag (32x16 f16): lane<16 -> col=lane, K={0..7,16..23}; lane>=16 -> col=lane-16, K={8..15,24..31}
// Beff is laid out so that each recurrent WG's slice is one contiguous 256KB block:
//   frag index = ((s*8 + w)*4 + g)*8 + kt   where nt = 16*g + s*8 + w
__global__ void k_pack_rec(const float* __restrict__ W_ih, const float* __restrict__ W_hh,
                           const float* __restrict__ W_lin, char* __restrict__ ws) {
  int tid = blockIdx.x * 256 + threadIdx.x;           // 16384 = 8kt * 64nt * 32lane
  int kt = tid >> 11, nt = (tid >> 5) & 63, lane = tid & 31;
  int nl = lane & 15, hi = lane >> 4;
  int n = nt * 16 + nl;
  _Float16* BH = (_Float16*)(ws + BHH_OFF);
  _Float16* BE = (_Float16*)(ws + BEFF_OFF);
  int g = nt >> 4, r = nt & 15, sp = r >> 3, wv = r & 7;
  size_t bhbase = ((size_t)(kt * 64 + nt)) * 512 + lane * 16;
  size_t bebase = ((size_t)(((sp * 8 + wv) * 4 + g) * 8 + kt)) * 512 + lane * 16;
  for (int i = 0; i < 16; ++i) {
    int k = kt * 32 + 8 * hi + i + (i >> 3) * 8;
    float whh = W_hh[(size_t)n * 256 + k];
    float d = 0.f;
    for (int f = 0; f < F_; ++f) d += W_lin[(size_t)f * 256 + k] * W_ih[(size_t)n * 320 + f];
    BH[bhbase + i] = (_Float16)whh;
    BE[bebase + i] = (_Float16)(whh + d);
  }
}

__global__ void k_pack_bx(const float* __restrict__ W_ih, char* __restrict__ ws) {
  int tid = blockIdx.x * 256 + threadIdx.x;
  int kt = tid >> 11, nt = (tid >> 5) & 63, lane = tid & 31;
  int nl = lane & 15, hi = lane >> 4;
  int n = nt * 16 + nl;
  _Float16* B = (_Float16*)(ws + BX_OFF);
  size_t base = ((size_t)(kt * 64 + nt)) * 512 + lane * 16;
  for (int i = 0; i < 16; ++i) {
    int k = kt * 32 + 8 * hi + i + (i >> 3) * 8;
    B[base + i] = (_Float16)W_ih[(size_t)n * 320 + F_ + k];
  }
}

__global__ void k_pack_bl(const float* __restrict__ W_lin, char* __restrict__ ws) {
  int tid = blockIdx.x * 256 + threadIdx.x;           // 1024 = 8kt * 4nt * 32lane
  int kt = tid >> 7, nt = (tid >> 5) & 3, lane = tid & 31;
  int nl = lane & 15, hi = lane >> 4;
  int n = nt * 16 + nl;
  _Float16* B = (_Float16*)(ws + BL_OFF);
  size_t base = ((size_t)(kt * 4 + nt)) * 512 + lane * 16;
  for (int i = 0; i < 16; ++i) {
    int k = kt * 32 + 8 * hi + i + (i >> 3) * 8;
    B[base + i] = (_Float16)W_lin[(size_t)n * 256 + k];
  }
}

// ---- Gx = x @ Wx.T + bsum : M=131072 N=1024 K=256, output in C-frag order ----
// 2 blocks per m-tile; each wave owns 4 n-tiles: 32 accum + 32 B-frag VGPRs,
// no spills, and a 8-load clause per k-slice hides L2 latency.
__global__ void k_gx(const float* __restrict__ X, char* __restrict__ ws) {
  int blk = blockIdx.x;                 // blk = mt*2 + half
  int mt = blk >> 1, half = blk & 1;    // mt = t*16 + bt
  int bt = mt & 15, t = mt >> 4;
  int tid = threadIdx.x;
  int w = tid >> 5, lane = tid & 31, nl = lane & 15, hi = lane >> 4;
  const float* bsum = (const float*)(ws + BSUM_OFF);
  char* GX = ws + GX_OFF;
  int b = bt * 16 + nl;
  const float* xrow = X + ((size_t)b * L_ + t) * H_;
  int nt0 = half * 32 + w * 4;
  f32x8 acc[4];
  for (int j = 0; j < 4; ++j) {
    float bs = bsum[(nt0 + j) * 16 + nl];
#pragma unroll
    for (int e = 0; e < 8; ++e) acc[j][e] = bs;
  }
  for (int kt = 0; kt < 8; ++kt) {
    Frag32 bf[4];
#pragma unroll
    for (int j = 0; j < 4; ++j) {
      const uint4* bp = (const uint4*)(ws + BX_OFF + ((size_t)(kt * 64 + nt0 + j)) * 1024 + lane * 32);
      bf[j].q[0] = bp[0]; bf[j].q[1] = bp[1];
    }
    Frag32 a;
    int k0 = kt * 32 + 8 * hi;        // A-frag: lane<16 row=nl K {0..7,16..23}; hi: {8..15,24..31}
#pragma unroll
    for (int i = 0; i < 8; ++i) a.h[i] = (_Float16)xrow[k0 + i];
#pragma unroll
    for (int i = 0; i < 8; ++i) a.h[8 + i] = (_Float16)xrow[k0 + 16 + i];
#pragma unroll
    for (int j = 0; j < 4; ++j)
      acc[j] = __builtin_amdgcn_wmma_f32_16x16x32_f16(false, a.v, false, bf[j].v, (short)0, acc[j], false, false);
  }
  for (int j = 0; j < 4; ++j) {
    Half8 o;
#pragma unroll
    for (int e = 0; e < 8; ++e) o.h[e] = (_Float16)acc[j][e];
    // Gx is stream-out, read exactly once much later: non-temporal store
    __builtin_nontemporal_store(o.p, (u32x4v*)(GX + ((size_t)(mt * 64 + nt0 + j)) * 512 + lane * 16));
  }
}

// ---- serial recurrence ---------------------------------------------------
__global__ void k_rec(const float* __restrict__ h0, const float* __restrict__ c0,
                      char* __restrict__ ws) {
  extern __shared__ char smem[];
  _Float16* Blds = (_Float16*)smem;                 // 256KB: Weff slice, frag order [w][g][kt]
  _Float16* hst  = (_Float16*)(smem + 262144);      // 16 x HPAD halfs: full h tile
  int bt = blockIdx.x, s = blockIdx.y;              // batch tile, N-split half
  int tid = threadIdx.x;
  int w = tid >> 5, lane = tid & 31, nl = lane & 15, hi = lane >> 4;
  int jt = s * 8 + w;                               // global j-tile owned by this wave
  int jcol = jt * 16 + nl;                          // hidden index in C-layout for this lane
  const char* BHH = ws + BHH_OFF;
  const char* GX  = ws + GX_OFF;
  const float* bias2 = (const float*)(ws + BIAS2_OFF);
  char* HALL = ws + HALL_OFF;
  unsigned* flags = (unsigned*)(ws + FLAGS_OFF);

  // Stage this WG's contiguous 256KB Weff slice into LDS with async global->LDS
  // DMA (no VGPR round-trip, tracked by ASYNCcnt). Identity offsets: LDS layout
  // == packed global layout. 16B per lane per op, coalesced.
  {
    const char* src = ws + BEFF_OFF + (size_t)s * 262144;
#pragma unroll 4
    for (int i = 0; i < 64; ++i) {
      unsigned off = (unsigned)(tid * 16 + i * 4096);
      unsigned lds_addr = off;                            // Blds starts at LDS offset 0
      unsigned long long ga = (unsigned long long)(src + off);
      asm volatile("global_load_async_to_lds_b128 %0, %1, off"
                   :: "v"(lds_addr), "v"(ga) : "memory");
    }
  }
  // h0 -> hstage (f16)
  for (int i = tid; i < 16 * H_; i += 256) {
    int row = i >> 8, col = i & 255;
    hst[row * HPAD + col] = (_Float16)h0[(size_t)(bt * 16 + row) * H_ + col];
  }
  // c state lives in registers, C-fragment layout
  f32x8 creg;
#pragma unroll
  for (int e = 0; e < 8; ++e)
    creg[e] = c0[(size_t)(bt * 16 + e + 8 * hi) * H_ + jcol];
  float b2v[4];
#pragma unroll
  for (int g = 0; g < 4; ++g) b2v[g] = bias2[g * H_ + jcol];
  asm volatile("s_wait_asynccnt 0x0" ::: "memory");   // weight DMA complete
  __syncthreads();

  for (int t = 0; t < L_; ++t) {
    f32x8 acc[4];
#pragma unroll
    for (int g = 0; g < 4; ++g) {
      int ntg = 16 * g + jt;
      Half8 gx;                                      // single-use stream-in: NT load
      gx.p = __builtin_nontemporal_load(
          (const u32x4v*)(GX + ((size_t)((t * 16 + bt) * 64 + ntg)) * 512 + lane * 16));
      float b2 = (t == 0) ? 0.f : b2v[g];
#pragma unroll
      for (int e = 0; e < 8; ++e) acc[g][e] = (float)gx.h[e] + b2;
      if (t + 1 < L_)
        __builtin_prefetch(GX + ((size_t)(((t + 1) * 16 + bt) * 64 + ntg)) * 512 + lane * 16, 0, 1);
    }
#pragma unroll 2
    for (int kt = 0; kt < 8; ++kt) {
      Frag32 a;
      int ka = kt * 32 + 8 * hi;
      a.q[0] = *(const uint4*)&hst[nl * HPAD + ka];
      a.q[1] = *(const uint4*)&hst[nl * HPAD + ka + 16];
      Frag32 bf[4];
      if (t == 0) {                                    // step 0: plain W_hh from L2
#pragma unroll
        for (int g = 0; g < 4; ++g) {
          int ntg = 16 * g + jt;
          const uint4* bp = (const uint4*)(BHH + ((size_t)(kt * 64 + ntg)) * 1024 + lane * 32);
          bf[g].q[0] = bp[0]; bf[g].q[1] = bp[1];
        }
      } else {                                         // t>=1: folded Weff from LDS
#pragma unroll
        for (int g = 0; g < 4; ++g) {
          const uint4* bp = (const uint4*)(Blds + ((size_t)((w * 4 + g) * 8 + kt)) * 512) + lane * 2;
          bf[g].q[0] = bp[0]; bf[g].q[1] = bp[1];
        }
      }
#pragma unroll
      for (int g = 0; g < 4; ++g)
        acc[g] = __builtin_amdgcn_wmma_f32_16x16x32_f16(false, a.v, false, bf[g].v, (short)0, acc[g], false, false);
    }
    __syncthreads();                                   // all hstage reads done
    // gate nonlinearity + state update, all in registers (this wave owns i/f/g/o of its j's)
#pragma unroll
    for (int e = 0; e < 8; ++e) {
      float iv = sigm_(acc[0][e]);
      float fv = sigm_(acc[1][e]);
      float gv = tanh_(acc[2][e]);
      float ov = sigm_(acc[3][e]);
      float cn = fv * creg[e] + iv * gv;
      creg[e] = cn;
      float hv = ov * tanh_(cn);
      hst[(e + 8 * hi) * HPAD + jcol] = (_Float16)hv;
    }
    __syncthreads();                                   // own half of h_new in LDS
    {
      int row = tid >> 4, ch = tid & 15;               // coalesced own-half -> h_all (4KB)
      uint4 v = *(const uint4*)&hst[row * HPAD + s * 128 + ch * 8];
      *(uint4*)(HALL + ((size_t)(t * B_ + bt * 16 + row)) * 512 + s * 256 + ch * 16) = v;
    }
    __threadfence();
    __syncthreads();
    if (tid == 0) {                                    // pairwise handshake with partner WG
      __hip_atomic_store(&flags[bt * 2 + s], (unsigned)(t + 1),
                         __ATOMIC_RELEASE, __HIP_MEMORY_SCOPE_AGENT);
      while (__hip_atomic_load(&flags[bt * 2 + (1 - s)],
                               __ATOMIC_ACQUIRE, __HIP_MEMORY_SCOPE_AGENT) <= (unsigned)t)
        __builtin_amdgcn_s_sleep(2);
    }
    __syncthreads();
    {
      int row = tid >> 4, ch = tid & 15;               // partner half L2 -> LDS (keep RT: must hit L2)
      int sp = 1 - s;
      uint4 v = *(const uint4*)(HALL + ((size_t)(t * B_ + bt * 16 + row)) * 512 + sp * 256 + ch * 16);
      *(uint4*)&hst[row * HPAD + sp * 128 + ch * 8] = v;
    }
    __syncthreads();
  }
}

// ---- out = h_all @ W_lin.T + b_lin : M=131072 N=64 K=256 ----------------
__global__ void k_proj(const float* __restrict__ b_lin, const char* __restrict__ ws,
                       float* __restrict__ out) {
  int id = blockIdx.x * 8 + (threadIdx.x >> 5);       // wave task
  int lane = threadIdx.x & 31, nl = lane & 15, hi = lane >> 4;
  int mt = id >> 2, nt = id & 3;
  const char* HALL = ws + HALL_OFF;
  const char* arow = HALL + (size_t)(mt * 16 + nl) * 512;
  f32x8 acc;
#pragma unroll
  for (int e = 0; e < 8; ++e) acc[e] = 0.f;
  for (int kt = 0; kt < 8; ++kt) {
    Frag32 a;
    int k0 = kt * 32 + 8 * hi;                        // byte offset = k0*2
    a.p[0] = __builtin_nontemporal_load((const u32x4v*)(arow + k0 * 2));        // last-use stream
    a.p[1] = __builtin_nontemporal_load((const u32x4v*)(arow + k0 * 2 + 32));
    Frag32 bf;
    const uint4* bp = (const uint4*)(ws + BL_OFF + ((size_t)(kt * 4 + nt)) * 1024 + lane * 32);
    bf.q[0] = bp[0]; bf.q[1] = bp[1];
    acc = __builtin_amdgcn_wmma_f32_16x16x32_f16(false, a.v, false, bf.v, (short)0, acc, false, false);
  }
  float bl = b_lin[nt * 16 + nl];
#pragma unroll
  for (int e = 0; e < 8; ++e) {
    int row = mt * 16 + e + 8 * hi;                   // row = t*256 + b
    int b = row & 255, t = row >> 8;
    out[((size_t)b * L_ + t) * F_ + nt * 16 + nl] = acc[e] + bl;
  }
}

extern "C" void kernel_launch(void* const* d_in, const int* in_sizes, int n_in,
                              void* d_out, int out_size, void* d_ws, size_t ws_size,
                              hipStream_t stream) {
  (void)in_sizes; (void)n_in; (void)out_size; (void)ws_size;
  const float* X     = (const float*)d_in[0];
  const float* h0    = (const float*)d_in[1];
  const float* c0    = (const float*)d_in[2];
  const float* W_ih  = (const float*)d_in[4];
  const float* W_hh  = (const float*)d_in[5];
  const float* b_ih  = (const float*)d_in[6];
  const float* b_hh  = (const float*)d_in[7];
  const float* W_lin = (const float*)d_in[8];
  const float* b_lin = (const float*)d_in[9];
  char* ws = (char*)d_ws;
  float* out = (float*)d_out;

  k_prep<<<4, 256, 0, stream>>>(b_ih, b_hh, b_lin, W_ih, ws);
  k_pack_rec<<<64, 256, 0, stream>>>(W_ih, W_hh, W_lin, ws);
  k_pack_bx<<<64, 256, 0, stream>>>(W_ih, ws);
  k_pack_bl<<<4, 256, 0, stream>>>(W_lin, ws);
  k_gx<<<16384, 256, 0, stream>>>(X, ws);
  (void)hipFuncSetAttribute(reinterpret_cast<const void*>(k_rec),
                            hipFuncAttributeMaxDynamicSharedMemorySize, REC_LDS);
  k_rec<<<dim3(16, 2), 256, REC_LDS, stream>>>(h0, c0, ws);
  k_proj<<<4096, 256, 0, stream>>>(b_lin, ws, out);
}
